// NaiveSDPA_4432406249990
// MI455X (gfx1250) — matrix operand order repaired and checked by reading, not compiled, not run on hardware
//
#include <hip/hip_runtime.h>
#include <hip/hip_bf16.h>

// SDPA flash-attention for gfx1250 (MI455X), transposed bf16 WMMA formulation.
// S^T = K * Q^T, softmax over keys per lane (lane = query), O^T = V^T * P^T.
// Register-pipelined staging: next chunk's global loads are in flight during
// the current chunk's WMMA/softmax. exp2-domain softmax (log2e folded into
// the Q scale and the mask FMA). N=16, S=SKV=4096, K=V=64, fp32 in/out.

#define N_B   16
#define S_Q   4096
#define S_KV  4096
#define DK    64
#define DV    64
#define NWAVE 8          // waves per block
#define KVCH  64         // keys per chunk (4 key tiles of 16)

typedef __attribute__((ext_vector_type(16))) __bf16 v16bf;
typedef __attribute__((ext_vector_type(8)))  float  v8f;

union Frag { unsigned u[8]; v16bf v; };

static __device__ __forceinline__ unsigned pack_bf16(float a, float b) {
    union { float f; unsigned u; } x, y; x.f = a; y.f = b;
    unsigned ra = x.u + 0x7fffu + ((x.u >> 16) & 1u);   // RNE
    unsigned rb = y.u + 0x7fffu + ((y.u >> 16) & 1u);
    return (ra >> 16) | (rb & 0xffff0000u);
}

__global__ __launch_bounds__(32 * NWAVE)
void sdpa_fwd_kernel(const float* __restrict__ queries,
                     const float* __restrict__ keys,
                     const float* __restrict__ values,
                     const float* __restrict__ mask,
                     float* __restrict__ out) {
    // K chunk row-major bf16 [key][feat]; V chunk transposed bf16 [feat][key].
    __shared__ __align__(16) unsigned short kbuf[KVCH][DK + 2];
    __shared__ __align__(16) unsigned short vtb [DV][KVCH + 2];

    const int tid    = threadIdx.x;
    const int wave   = tid >> 5;
    const int lane   = tid & 31;
    const int lane15 = lane & 15;      // this lane's query within the tile
    const int h      = lane >> 4;      // lane half

    const int b      = blockIdx.y;
    const int qrow0  = blockIdx.x * (16 * NWAVE) + wave * 16;
    const int q      = qrow0 + lane15; // global query row for this lane

    const float LOG2E  = 1.44269504088896340736f;
    const float qscale = 0.125f * LOG2E;   // 1/sqrt(64) * log2(e)
    const float* qb = queries + (size_t)b * S_Q  * DK;
    const float* kb = keys    + (size_t)b * S_KV * DK;
    const float* vb = values  + (size_t)b * S_KV * DV;
    const float* mptr = mask + (size_t)q * S_KV;

    // Per-thread staging assignment (uniform across the loop):
    //   K: 4 float4 units, unit i -> key row rK[i], feature col cK[i].
    //   V: 2 row-pair units, unit u -> rows 2*rpV[u], 2*rpV[u]+1, col cV[u].
    int rK[4], cK[4];
#pragma unroll
    for (int i = 0; i < 4; ++i) {
        int e = tid + i * 256;
        rK[i] = e >> 4;
        cK[i] = (e & 15) << 2;
    }
    int rpV[2], cV[2];
#pragma unroll
    for (int u = 0; u < 2; ++u) {
        int e = tid + u * 256;
        rpV[u] = e >> 4;               // 0..31 row pairs
        cV[u]  = (e & 15) << 2;
    }

    // ---- Q^T as two resident bf16 B-fragments (feat x query), pre-scaled ----
    // B layout: lane n = query, VGPR v holds feature pair k = s*32 + h*16 + 2v.
    Frag bq[2];
#pragma unroll
    for (int s = 0; s < 2; ++s) {
#pragma unroll
        for (int v = 0; v < 8; ++v) {
            int k = s * 32 + h * 16 + 2 * v;
            const float2 q2 = *(const float2*)(qb + (size_t)q * DK + k);
            bq[s].u[v] = pack_bf16(q2.x * qscale, q2.y * qscale);
        }
    }

    // ---- per-lane online-softmax state (query-scalar) + O^T accumulators ----
    float mrun = -1e30f, lrun = 0.0f;
    v8f o[4];                          // O^T: fragment f -> feats f*16 + v + 8h
#pragma unroll
    for (int f = 0; f < 4; ++f) o[f] = {};

    // ---- pipelined preload of chunk 0 ----
    float4 kreg[4], vregA[2], vregB[2];
#pragma unroll
    for (int i = 0; i < 4; ++i)
        kreg[i] = *(const float4*)(kb + (size_t)rK[i] * DK + cK[i]);
#pragma unroll
    for (int u = 0; u < 2; ++u) {
        vregA[u] = *(const float4*)(vb + (size_t)(2 * rpV[u])     * DV + cV[u]);
        vregB[u] = *(const float4*)(vb + (size_t)(2 * rpV[u] + 1) * DV + cV[u]);
    }

    for (int kv = 0; kv < S_KV; kv += KVCH) {
        __syncthreads();   // previous chunk fully consumed

        // ---- convert preloaded registers -> LDS (bf16, dword-packed) ----
#pragma unroll
        for (int i = 0; i < 4; ++i) {
            *(unsigned*)&kbuf[rK[i]][cK[i]]     = pack_bf16(kreg[i].x, kreg[i].y);
            *(unsigned*)&kbuf[rK[i]][cK[i] + 2] = pack_bf16(kreg[i].z, kreg[i].w);
        }
#pragma unroll
        for (int u = 0; u < 2; ++u) {
            const float* pa = (const float*)&vregA[u];
            const float* pb = (const float*)&vregB[u];
#pragma unroll
            for (int j = 0; j < 4; ++j)   // pairs (key 2rp, key 2rp+1)
                *(unsigned*)&vtb[cV[u] + j][2 * rpV[u]] = pack_bf16(pa[j], pb[j]);
        }
        __syncthreads();

        // ---- kick off next chunk's global loads (covered by compute) ----
        const int kvn = kv + KVCH;
        if (kvn < S_KV) {
#pragma unroll
            for (int i = 0; i < 4; ++i)
                kreg[i] = *(const float4*)(kb + (size_t)(kvn + rK[i]) * DK + cK[i]);
#pragma unroll
            for (int u = 0; u < 2; ++u) {
                vregA[u] = *(const float4*)(vb + (size_t)(kvn + 2 * rpV[u])     * DV + cV[u]);
                vregB[u] = *(const float4*)(vb + (size_t)(kvn + 2 * rpV[u] + 1) * DV + cV[u]);
            }
        }

        // ---- S^T = K * Q^T : 4 key tiles x 2 k-steps of WMMA ----
        // A layout (16x32): lane m = key = kt*16+lane15, VGPR v holds feature
        // pair k0 = h*16 + 2v  (+32 for second step).
        v8f sc[4];
#pragma unroll
        for (int kt = 0; kt < 4; ++kt) {
            Frag a0, a1;
            int key = kt * 16 + lane15;
#pragma unroll
            for (int v = 0; v < 8; ++v) {
                int off = h * 16 + 2 * v;
                a0.u[v] = *(const unsigned*)&kbuf[key][off];
                a1.u[v] = *(const unsigned*)&kbuf[key][32 + off];
            }
            v8f z = {};
            v8f t = __builtin_amdgcn_wmma_f32_16x16x32_bf16(
                false, a0.v, false, bq[0].v, (short)0, z, false, false);
            sc[kt] = __builtin_amdgcn_wmma_f32_16x16x32_bf16(
                false, a1.v, false, bq[1].v, (short)0, t, false, false);
        }

        // ---- add mask * log2e (FMA); lane's 8 elements are consecutive cols ----
        // S^T C-layout: row = key = kt*16 + v + 8h, col = query = lane15.
#pragma unroll
        for (int kt = 0; kt < 4; ++kt) {
            const float4 mA = *(const float4*)(mptr + kv + kt * 16 + 8 * h);
            const float4 mB = *(const float4*)(mptr + kv + kt * 16 + 8 * h + 4);
            sc[kt][0] = fmaf(mA.x, LOG2E, sc[kt][0]);
            sc[kt][1] = fmaf(mA.y, LOG2E, sc[kt][1]);
            sc[kt][2] = fmaf(mA.z, LOG2E, sc[kt][2]);
            sc[kt][3] = fmaf(mA.w, LOG2E, sc[kt][3]);
            sc[kt][4] = fmaf(mB.x, LOG2E, sc[kt][4]);
            sc[kt][5] = fmaf(mB.y, LOG2E, sc[kt][5]);
            sc[kt][6] = fmaf(mB.z, LOG2E, sc[kt][6]);
            sc[kt][7] = fmaf(mB.w, LOG2E, sc[kt][7]);
        }

        // ---- online softmax (exp2 domain): local reduce + ONE shuffle each ----
        float mx = sc[0][0];
#pragma unroll
        for (int kt = 0; kt < 4; ++kt)
#pragma unroll
            for (int v = 0; v < 8; ++v) mx = fmaxf(mx, sc[kt][v]);
        mx = fmaxf(mx, __shfl_xor(mx, 16, 32));
        float mnew  = fmaxf(mrun, mx);
        float alpha = __builtin_amdgcn_exp2f(mrun - mnew);
        mrun = mnew;
        float rs = 0.0f;
#pragma unroll
        for (int kt = 0; kt < 4; ++kt)
#pragma unroll
            for (int v = 0; v < 8; ++v) {
                float p = __builtin_amdgcn_exp2f(sc[kt][v] - mnew);
                sc[kt][v] = p;
                rs += p;
            }
        rs += __shfl_xor(rs, 16, 32);
        lrun = lrun * alpha + rs;
#pragma unroll
        for (int f = 0; f < 4; ++f)
#pragma unroll
            for (int v = 0; v < 8; ++v) o[f][v] *= alpha;

        // ---- build P^T B-fragments from S^T C-fragments (half-swap only) ----
        // bp[s].u[v] needs keys j0 = s*32 + h*16 + 2v -> tile kt = 2s+h,
        // in-tile idx = 2v; idx<8 lives in half 0 lanes, idx>=8 in half 1.
        Frag bp[2];
#pragma unroll
        for (int s = 0; s < 2; ++s) {
            float plow[8], poth[8];      // my tile (2s+h) / partner's tile
#pragma unroll
            for (int v = 0; v < 8; ++v) {
                plow[v] = h ? sc[2 * s + 1][v] : sc[2 * s][v];
                poth[v] = h ? sc[2 * s][v]     : sc[2 * s + 1][v];
            }
#pragma unroll
            for (int j = 0; j < 4; ++j) {
                unsigned own = pack_bf16(plow[2 * j], plow[2 * j + 1]);
                unsigned snd = pack_bf16(poth[2 * j], poth[2 * j + 1]);
                unsigned rcv = (unsigned)__shfl_xor((int)snd, 16, 32);
                bp[s].u[j]     = h ? rcv : own;
                bp[s].u[4 + j] = h ? own : rcv;
            }
        }

        // ---- O^T += V^T * P^T : 4 feature tiles x 2 k-steps ----
#pragma unroll
        for (int f = 0; f < 4; ++f) {
            Frag av0, av1;
            int feat = f * 16 + lane15;
#pragma unroll
            for (int v = 0; v < 8; ++v) {
                int off = h * 16 + 2 * v;
                av0.u[v] = *(const unsigned*)&vtb[feat][off];
                av1.u[v] = *(const unsigned*)&vtb[feat][32 + off];
            }
            o[f] = __builtin_amdgcn_wmma_f32_16x16x32_bf16(
                false, av0.v, false, bp[0].v, (short)0, o[f], false, false);
            o[f] = __builtin_amdgcn_wmma_f32_16x16x32_bf16(
                false, av1.v, false, bp[1].v, (short)0, o[f], false, false);
        }
    }

    // ---- epilogue: normalize; lane's 8 feats per tile are consecutive ----
    float inv = 1.0f / lrun;
    float* orow = out + ((size_t)b * S_Q + q) * DV;
#pragma unroll
    for (int f = 0; f < 4; ++f) {
        float4 s0 = make_float4(o[f][0] * inv, o[f][1] * inv,
                                o[f][2] * inv, o[f][3] * inv);
        float4 s1 = make_float4(o[f][4] * inv, o[f][5] * inv,
                                o[f][6] * inv, o[f][7] * inv);
        *(float4*)(orow + f * 16 + 8 * h)     = s0;
        *(float4*)(orow + f * 16 + 8 * h + 4) = s1;
    }
}

extern "C" void kernel_launch(void* const* d_in, const int* in_sizes, int n_in,
                              void* d_out, int out_size, void* d_ws, size_t ws_size,
                              hipStream_t stream) {
    const float* queries = (const float*)d_in[0];
    const float* keys    = (const float*)d_in[1];
    const float* values  = (const float*)d_in[2];
    const float* mask    = (const float*)d_in[3];
    float* out = (float*)d_out;

    dim3 grid(S_Q / (16 * NWAVE), N_B);   // (32, 16)
    dim3 block(32 * NWAVE);               // 256 threads = 8 waves
    sdpa_fwd_kernel<<<grid, block, 0, stream>>>(queries, keys, values, mask, out);
}
